// Fsmm_rnn_13915694039054
// MI455X (gfx1250) — matrix operand order repaired and checked by reading, not compiled, not run on hardware
//
#include <hip/hip_runtime.h>
#include <hip/hip_bf16.h>
#include <math.h>

// Problem constants (from reference)
#define Bsz   64
#define Tlen  2048
#define Din   4
#define Hdim  256
#define WIN   64

typedef __attribute__((ext_vector_type(16))) _Float16 v16h;
typedef __attribute__((ext_vector_type(8)))  _Float16 v8h;
typedef __attribute__((ext_vector_type(8)))  float    v8f;

// Hardware tanh (V_TANH_F32 is a gfx1250 TRANS op) with safe fallback.
#if __has_builtin(__builtin_amdgcn_tanhf)
#define FAST_TANH(x) __builtin_amdgcn_tanhf(x)
#elif __has_builtin(__builtin_amdgcn_tanh_f32)
#define FAST_TANH(x) __builtin_amdgcn_tanh_f32(x)
#else
#define FAST_TANH(x) tanhf(x)
#endif

// ---------------------------------------------------------------------------
// Kernel 1: f32 -> f16 weight conversion (row-major layout kept: for the
// GEMM  H @ W^T  the WMMA B-fragment of column-tile n, k-slice is the
// two 8-wide runs W[n, k+8h ..], W[n, k+16+8h ..] (h = lane>>4), so no repacking is needed).
// ---------------------------------------------------------------------------
__global__ void convert_weights_kernel(const float* __restrict__ whh0,
                                       const float* __restrict__ wih1,
                                       const float* __restrict__ whh1,
                                       _Float16* __restrict__ wf16) {
  int i = blockIdx.x * blockDim.x + threadIdx.x;
  if (i < Hdim * Hdim) {
    wf16[i]                     = (_Float16)whh0[i];
    wf16[Hdim * Hdim + i]       = (_Float16)wih1[i];
    wf16[2 * Hdim * Hdim + i]   = (_Float16)whh1[i];
  }
}

// ---------------------------------------------------------------------------
// WMMA fragment loaders.
// A (16x32 f16): lane L holds row m=L&15; K values {kh*8+0..7} U {16+kh*8+0..7}
// with kh=L>>4  -> two contiguous 16B runs (2x ds_load_b128).
// B (32x16 f16): lane L holds col n=L&15, same K set as A -> two
// contiguous 16B runs of W[n, ...] in the row-major f16 weight copy.
// ---------------------------------------------------------------------------
__device__ __forceinline__ v16h ldA(const _Float16* p) {
  v8h lo = *(const v8h*)p;
  v8h hi = *(const v8h*)(p + 16);
  return __builtin_shufflevector(lo, hi, 0,1,2,3,4,5,6,7,8,9,10,11,12,13,14,15);
}
__device__ __forceinline__ v16h ldB(const _Float16* p) {
  v8h lo = *(const v8h*)p;
  v8h hi = *(const v8h*)(p + 16);
  return __builtin_shufflevector(lo, hi, 0,1,2,3,4,5,6,7,8,9,10,11,12,13,14,15);
}

// ---------------------------------------------------------------------------
// Kernel 2: persistent single-workgroup 2-layer RNN scan.
// 512 threads = 16 waves (wave32). Wave w owns hidden-column tile w (16 cols)
// and all 4 batch row-tiles. Hidden states live in LDS as f16, double
// buffered. The 8 Whh0 B-fragments are PINNED in registers (64 VGPRs, via
// opaque asm so they cannot be rematerialized) -> layer 0 is pure LDS+WMMA.
// Wih1/Whh1 are force-streamed from L2 each step (opaque pointers block
// LICM), keeping peak pressure ~170 VGPRs so the scheduler pipelines the
// per-group loads against the WMMA chain with graduated waits.
// ---------------------------------------------------------------------------
__global__ __launch_bounds__(512) void rnn_scan_kernel(
    const float* __restrict__ x,        // (B,T,D)
    const float* __restrict__ wih0,     // (H,D)
    const float* __restrict__ bih0, const float* __restrict__ bhh0,
    const float* __restrict__ bih1, const float* __restrict__ bhh1,
    const _Float16* __restrict__ wf16,  // whh0 | wih1 | whh1, each HxH f16
    float* __restrict__ ys1_out,        // (B,H) layer-0 state at t=T-1
    float* __restrict__ hT_out)         // (B,H) layer-1 state at t=T-1
{
  extern __shared__ char smem[];
  _Float16* h0buf = (_Float16*)smem;                       // [2][64][256]
  _Float16* h1buf = (_Float16*)(smem + 65536);             // [2][64][256]
  float*    xsh   = (float*)(smem + 131072);               // [64][4]
  float*    wih0s = (float*)(smem + 132096);               // [256][4]
  float*    b0s   = (float*)(smem + 136192);               // [256]
  float*    b1s   = (float*)(smem + 137216);               // [256]

  const int tid  = threadIdx.x;
  const int lane = tid & 31;
  const int wv   = tid >> 5;           // wave id 0..15  == hidden col tile
  const int nloc = lane & 15;
  const int kh   = lane >> 4;          // K half selector
  const int hcol = wv * 16 + nloc;     // this lane's output hidden column
  const int mrb  = 8 * kh;             // C/D layout: M = v + 8*(lane>>4)

  const _Float16* Wh0 = wf16;
  const _Float16* Wi1 = wf16 + Hdim * Hdim;
  const _Float16* Wh1 = wf16 + 2 * Hdim * Hdim;

  // init LDS: zero state buffer 0, stage Wih0 + combined biases
  for (int i = tid; i < Bsz * Hdim; i += 512) {
    h0buf[i] = (_Float16)0.0f;
    h1buf[i] = (_Float16)0.0f;
  }
  for (int i = tid; i < Hdim * Din; i += 512) wih0s[i] = wih0[i];
  for (int i = tid; i < Hdim; i += 512) {
    b0s[i] = bih0[i] + bhh0[i];
    b1s[i] = bih1[i] + bhh1[i];
  }
  __syncthreads();

  // per-lane constants: input-projection row for this lane's hidden column
  const float w0 = wih0s[hcol * 4 + 0];
  const float w1 = wih0s[hcol * 4 + 1];
  const float w2 = wih0s[hcol * 4 + 2];
  const float w3 = wih0s[hcol * 4 + 3];
  const float b0c = b0s[hcol];
  const float b1c = b1s[hcol];

  const size_t wrow = (size_t)hcol * Hdim + kh * 8;  // B-fragment row base

  // Pin the recurrent layer-0 weights in registers for the whole scan:
  // 8 fragments x 8 VGPRs = 64 VGPRs. The opaque asm makes each fragment's
  // definition non-rematerializable, so layer 0 runs with ZERO global loads.
  v16h wh0f[8];
#pragma unroll
  for (int kt = 0; kt < 8; ++kt) {
    wh0f[kt] = ldB(Wh0 + wrow + kt * 32);
    asm volatile("" : "+v"(wh0f[kt]));
  }

  for (int t = 0; t < Tlen; ++t) {
    const int cur = t & 1, nxt = cur ^ 1;
    _Float16* h0c = h0buf + cur * Bsz * Hdim;
    _Float16* h0n = h0buf + nxt * Bsz * Hdim;
    _Float16* h1c = h1buf + cur * Bsz * Hdim;
    _Float16* h1n = h1buf + nxt * Bsz * Hdim;

    // Opaque copies: block LICM so Wih1/Whh1 fragments are re-streamed from
    // L2 each iteration instead of pinning 128 more VGPRs.
    const _Float16* Wi1t = Wi1;
    const _Float16* Wh1t = Wh1;
    asm volatile("" : "+v"(Wi1t), "+v"(Wh1t));

    __syncthreads();                       // prev step fully visible, xsh free
    if (tid < Bsz * Din) {
      int b = tid >> 2, d = tid & 3;
      xsh[tid] = x[b * (Tlen * Din) + t * Din + d];
    }
    // Warm the streamed layer-1 weights (L2 -> near cache) while layer 0 runs.
#pragma unroll
    for (int kt = 0; kt < 8; ++kt) {
      __builtin_prefetch(Wi1t + wrow + kt * 32, 0, 3);
      __builtin_prefetch(Wh1t + wrow + kt * 32, 0, 3);
    }
    __syncthreads();

    // -------- layer 0: h0' = tanh(x W0^T + h0 Whh0^T + b) --------
    v8f acc[4];
#pragma unroll
    for (int tm = 0; tm < 4; ++tm) {
#pragma unroll
      for (int v = 0; v < 8; ++v) {
        int b = tm * 16 + mrb + v;
        const float* xb = xsh + b * 4;
        acc[tm][v] = b0c + xb[0] * w0 + xb[1] * w1 + xb[2] * w2 + xb[3] * w3;
      }
    }
#pragma unroll
    for (int kt = 0; kt < 8; ++kt) {
      const int kb = kt * 32;
      // batch the 4 independent A loads ahead of the 4 WMMAs
      v16h a0 = ldA(h0c + (0 * 16 + nloc) * Hdim + kb + kh * 8);
      v16h a1 = ldA(h0c + (1 * 16 + nloc) * Hdim + kb + kh * 8);
      v16h a2 = ldA(h0c + (2 * 16 + nloc) * Hdim + kb + kh * 8);
      v16h a3 = ldA(h0c + (3 * 16 + nloc) * Hdim + kb + kh * 8);
      acc[0] = __builtin_amdgcn_wmma_f32_16x16x32_f16(false, a0, false, wh0f[kt],
                                                      (short)0, acc[0], false, false);
      acc[1] = __builtin_amdgcn_wmma_f32_16x16x32_f16(false, a1, false, wh0f[kt],
                                                      (short)0, acc[1], false, false);
      acc[2] = __builtin_amdgcn_wmma_f32_16x16x32_f16(false, a2, false, wh0f[kt],
                                                      (short)0, acc[2], false, false);
      acc[3] = __builtin_amdgcn_wmma_f32_16x16x32_f16(false, a3, false, wh0f[kt],
                                                      (short)0, acc[3], false, false);
    }
#pragma unroll
    for (int tm = 0; tm < 4; ++tm)
#pragma unroll
      for (int v = 0; v < 8; ++v) {
        int b = tm * 16 + mrb + v;
        h0n[b * Hdim + hcol] = (_Float16)FAST_TANH(acc[tm][v]);
      }
    __syncthreads();

    // -------- layer 1: h1' = tanh(h0' Wih1^T + h1 Whh1^T + b) --------
#pragma unroll
    for (int tm = 0; tm < 4; ++tm)
#pragma unroll
      for (int v = 0; v < 8; ++v) acc[tm][v] = b1c;
#pragma unroll
    for (int kt = 0; kt < 8; ++kt) {
      const int kb = kt * 32;
      v16h bfW = ldB(Wi1t + wrow + kb);   // streamed (L2)
      v16h bfU = ldB(Wh1t + wrow + kb);   // streamed (L2)
      // process batch tiles in pairs: 4 independent loads, then 4 WMMAs
      {
        v16h a00 = ldA(h0n + (0 * 16 + nloc) * Hdim + kb + kh * 8);
        v16h a10 = ldA(h1c + (0 * 16 + nloc) * Hdim + kb + kh * 8);
        v16h a01 = ldA(h0n + (1 * 16 + nloc) * Hdim + kb + kh * 8);
        v16h a11 = ldA(h1c + (1 * 16 + nloc) * Hdim + kb + kh * 8);
        acc[0] = __builtin_amdgcn_wmma_f32_16x16x32_f16(false, a00, false, bfW,
                                                        (short)0, acc[0], false, false);
        acc[0] = __builtin_amdgcn_wmma_f32_16x16x32_f16(false, a10, false, bfU,
                                                        (short)0, acc[0], false, false);
        acc[1] = __builtin_amdgcn_wmma_f32_16x16x32_f16(false, a01, false, bfW,
                                                        (short)0, acc[1], false, false);
        acc[1] = __builtin_amdgcn_wmma_f32_16x16x32_f16(false, a11, false, bfU,
                                                        (short)0, acc[1], false, false);
      }
      {
        v16h a02 = ldA(h0n + (2 * 16 + nloc) * Hdim + kb + kh * 8);
        v16h a12 = ldA(h1c + (2 * 16 + nloc) * Hdim + kb + kh * 8);
        v16h a03 = ldA(h0n + (3 * 16 + nloc) * Hdim + kb + kh * 8);
        v16h a13 = ldA(h1c + (3 * 16 + nloc) * Hdim + kb + kh * 8);
        acc[2] = __builtin_amdgcn_wmma_f32_16x16x32_f16(false, a02, false, bfW,
                                                        (short)0, acc[2], false, false);
        acc[2] = __builtin_amdgcn_wmma_f32_16x16x32_f16(false, a12, false, bfU,
                                                        (short)0, acc[2], false, false);
        acc[3] = __builtin_amdgcn_wmma_f32_16x16x32_f16(false, a03, false, bfW,
                                                        (short)0, acc[3], false, false);
        acc[3] = __builtin_amdgcn_wmma_f32_16x16x32_f16(false, a13, false, bfU,
                                                        (short)0, acc[3], false, false);
      }
    }
#pragma unroll
    for (int tm = 0; tm < 4; ++tm)
#pragma unroll
      for (int v = 0; v < 8; ++v) {
        int b = tm * 16 + mrb + v;
        h1n[b * Hdim + hcol] = (_Float16)FAST_TANH(acc[tm][v]);
      }
  }
  __syncthreads();
  // T even -> final states sit in buffer 0
  for (int i = tid; i < Bsz * Hdim; i += 512) {
    ys1_out[i] = (float)h0buf[i];
    hT_out[i]  = (float)h1buf[i];
  }
}

// ---------------------------------------------------------------------------
// Kernel 3: per-batch epilogue. One block per batch element.
// Computes soc, last-step backward d_soc, f_soc affine scalars, Up[:,-1].
// ---------------------------------------------------------------------------
__device__ __forceinline__ float blockReduce256(float* red, float v) {
  int h = threadIdx.x;
  red[h] = v;
  __syncthreads();
  for (int s = 128; s > 0; s >>= 1) {
    if (h < s) red[h] += red[h + s];
    __syncthreads();
  }
  float r = red[0];
  __syncthreads();
  return r;
}

__global__ __launch_bounds__(256) void epilogue_batch_kernel(
    const float* __restrict__ x,
    const float* __restrict__ wih0, const float* __restrict__ wih1,
    const float* __restrict__ fcW,  const float* __restrict__ fcb,
    const float* __restrict__ f1W,  const float* __restrict__ f1b,
    const float* __restrict__ f2W,  const float* __restrict__ f2b,
    const float* __restrict__ R1p,  const float* __restrict__ C1p,
    const float* __restrict__ binom,
    const float* __restrict__ ys1d, const float* __restrict__ hTd,
    float* __restrict__ scal)       // [64][8]: soc,dsoc,fsoc,Up,v,I
{
  __shared__ float red[256];
  __shared__ float g1s[256];
  const int b = blockIdx.x;
  const int h = threadIdx.x;

  float hT = hTd[b * Hdim + h];
  float fw = fcW[h];

  // soc[b] = hT . fcW + fcb
  float soc_b = blockReduce256(red, hT * fw) + fcb[0];

  // backward through last cell of each layer
  g1s[h] = fw * (1.0f - hT * hT);
  __syncthreads();
  float s = 0.0f;
  for (int hh = 0; hh < Hdim; ++hh) s += g1s[hh] * wih1[hh * Hdim + h];
  float ys = ys1d[b * Hdim + h];
  float g0 = s * (1.0f - ys * ys);
  float dsoc = blockReduce256(red, g0 * wih0[h * Din + 3]);

  // f_soc = soc * (f2W.f1W) + (f2W.f1b + f2b)
  float s1 = blockReduce256(red, f2W[h] * f1W[h]);
  float s0 = blockReduce256(red, f2W[h] * f1b[h]);

  if (h == 0) {
    const float* xb = x + (size_t)b * Tlen * Din;
    float v_last = xb[(Tlen - 1) * Din + 0];
    float I_last = xb[(Tlen - 1) * Din + 1];
    // hist[T-1] = sum_k binom[k+1] * v[T-WIN+k]
    float hist = 0.0f;
    for (int k = 0; k < WIN; ++k)
      hist += binom[k + 1] * xb[(Tlen - WIN + k) * Din + 0];
    // Ts = ((time[T-1]-time[T-WIN]) / (WIN-1)) ** 0.5
    float tdiff = (xb[(Tlen - 1) * Din + 3] - xb[(Tlen - WIN) * Din + 3]) /
                  (float)(WIN - 1);
    float Ts = powf(tdiff, 0.5f);
    float R1 = R1p[0], C1 = C1p[0];
    float Up = -Ts / (R1 * C1) * v_last + Ts / C1 * I_last - hist;

    float* sc = scal + b * 8;
    sc[0] = soc_b;
    sc[1] = dsoc;
    sc[2] = soc_b * s1 + s0 + f2b[0];
    sc[3] = Up;
    sc[4] = v_last;
    sc[5] = I_last;
  }
}

// ---------------------------------------------------------------------------
// Kernel 4: assemble outputs: soc (64) | loss1 (64x64) | loss2 (64x64).
// loss1[i,j] = delta/Q * I_last[i] + dsoc[j]   ((64,1)+(64,) broadcast)
// loss2[i,j] = fsoc[i] - v_last[i] - R0*I_last[i] - Up[j]
// ---------------------------------------------------------------------------
__global__ void finalize_kernel(const float* __restrict__ scal,
                                const float* __restrict__ Qp,
                                const float* __restrict__ deltap,
                                const float* __restrict__ R0p,
                                float* __restrict__ out) {
  int idx = blockIdx.x * blockDim.x + threadIdx.x;
  if (idx < Bsz) {
    out[idx] = scal[idx * 8 + 0];
  } else if (idx < Bsz + Bsz * Bsz) {
    int k = idx - Bsz, i = k >> 6, j = k & 63;
    out[idx] = (deltap[0] / Qp[0]) * scal[i * 8 + 5] + scal[j * 8 + 1];
  } else if (idx < Bsz + 2 * Bsz * Bsz) {
    int k = idx - Bsz - Bsz * Bsz, i = k >> 6, j = k & 63;
    out[idx] = scal[i * 8 + 2] - scal[i * 8 + 4] -
               R0p[0] * scal[i * 8 + 5] - scal[j * 8 + 3];
  }
}

// ---------------------------------------------------------------------------
extern "C" void kernel_launch(void* const* d_in, const int* in_sizes, int n_in,
                              void* d_out, int out_size, void* d_ws, size_t ws_size,
                              hipStream_t stream) {
  const float* x     = (const float*)d_in[0];
  const float* Wih0  = (const float*)d_in[1];
  const float* Whh0  = (const float*)d_in[2];
  const float* bih0  = (const float*)d_in[3];
  const float* bhh0  = (const float*)d_in[4];
  const float* Wih1  = (const float*)d_in[5];
  const float* Whh1  = (const float*)d_in[6];
  const float* bih1  = (const float*)d_in[7];
  const float* bhh1  = (const float*)d_in[8];
  const float* fcW   = (const float*)d_in[9];
  const float* fcb   = (const float*)d_in[10];
  const float* Q     = (const float*)d_in[11];
  const float* delta = (const float*)d_in[12];
  const float* f1W   = (const float*)d_in[13];
  const float* f1b   = (const float*)d_in[14];
  const float* f2W   = (const float*)d_in[15];
  const float* f2b   = (const float*)d_in[16];
  // d_in[17] = U0 (unused by reference outputs)
  const float* R0    = (const float*)d_in[18];
  const float* C1    = (const float*)d_in[19];
  const float* R1    = (const float*)d_in[20];
  const float* binom = (const float*)d_in[21];

  char* ws = (char*)d_ws;
  _Float16* wf16 = (_Float16*)ws;                    // 3 * 65536 f16 = 384 KB
  float* ys1d = (float*)(ws + 393216);               // 64x256 f32
  float* hTd  = (float*)(ws + 458752);               // 64x256 f32
  float* scal = (float*)(ws + 524288);               // 64x8 f32

  convert_weights_kernel<<<(Hdim * Hdim + 255) / 256, 256, 0, stream>>>(
      Whh0, Wih1, Whh1, wf16);

  const size_t lds_bytes = 138240;  // 2x(2x64x256 f16) + x/wih0/bias staging
  rnn_scan_kernel<<<1, 512, lds_bytes, stream>>>(
      x, Wih0, bih0, bhh0, bih1, bhh1, wf16, ys1d, hTd);

  epilogue_batch_kernel<<<Bsz, 256, 0, stream>>>(
      x, Wih0, Wih1, fcW, fcb, f1W, f1b, f2W, f2b, R1, C1, binom,
      ys1d, hTd, scal);

  const int nout = Bsz + 2 * Bsz * Bsz;  // 8256
  finalize_kernel<<<(nout + 255) / 256, 256, 0, stream>>>(
      scal, Q, delta, R0, (float*)d_out);
}
